// SHGN_62534723830283
// MI455X (gfx1250) — matrix operand order repaired and checked by reading, not compiled, not run on hardware
//
#include <hip/hip_runtime.h>
#include <hip/hip_bf16.h>

#define N_NODES 50000
#define E_EDGES 500000
#define FD 128
#define VN 12
#define TN 768
#define REL 200
#define NET 3
#define NIDX 10000

typedef __attribute__((ext_vector_type(16))) _Float16 v16h;
typedef __attribute__((ext_vector_type(8)))  _Float16 v8h;
typedef __attribute__((ext_vector_type(8)))  float    v8f;
typedef __attribute__((ext_vector_type(4)))  float    v4f;

#define BS_STRIDE 40   // halves per LDS row (32 data + 8 pad), 80 B, 16-B aligned

// ---------------------------------------------------------------------------
// order-preserving float <-> uint keys for atomicMax-based segment_max
// ---------------------------------------------------------------------------
__device__ __forceinline__ unsigned f2key(float f) {
    unsigned b = __float_as_uint(f);
    return (b & 0x80000000u) ? ~b : (b | 0x80000000u);
}
__device__ __forceinline__ float key2f(unsigned k) {
    return (k & 0x80000000u) ? __uint_as_float(k & 0x7FFFFFFFu)
                             : __uint_as_float(~k);
}

// ---------------------------------------------------------------------------
// WMMA GEMM: C[M x 128] = act(A[M x K] @ B[K x 128] + bias),  K % 32 == 0.
// f16 multiplies, f32 accumulate via v_wmma_f32_16x16x32_f16.
// B tile staged in LDS column-major -> each lane's fragment is two 16-B runs 32 B
// apart (2x ds_load_b128); all 8 fragments loaded before the 8 WMMAs so DS
// latency overlaps matrix issue. A fragment = 4x global_load_b128 + cvt_pk.
// act: 0 = none, 1 = leaky_relu(0.01)
// ---------------------------------------------------------------------------
__global__ __launch_bounds__(256)
void shgn_gemm_x128(const float* __restrict__ A, int lda, int M, int K,
                    const float* __restrict__ B, const float* __restrict__ bias,
                    float* __restrict__ C, int ldc, int coloff, int act)
{
    __shared__ _Float16 Bs[FD][BS_STRIDE];   // [col][k] (transposed tile)

    const int tid  = threadIdx.x;
    const int wave = tid >> 5;
    const int lane = tid & 31;
    const int half = lane >> 4;       // 0: lanes 0-15, 1: lanes 16-31
    const int lrow = lane & 15;
    const int rowbase = blockIdx.x * 128 + wave * 16;
    const int arow = rowbase + lrow;  // A-fragment: lane holds row M = lane%16
    const bool rowok = arow < M;
    const float* Arow = A + (size_t)(rowok ? arow : 0) * lda;

    // staging coords: this thread owns column sn, k-range [sk, sk+16)
    const int sn = tid & 127;
    const int sk = (tid >> 7) * 16;

    v8f acc[8] = {};

    for (int k0 = 0; k0 < K; k0 += 32) {
        // ---- stage B[k0..k0+32) x 128 into LDS (f16, column-major) ----
        {
            v8h lo, hi;
            #pragma unroll
            for (int kk = 0; kk < 8; ++kk) {
                lo[kk] = (_Float16)B[(size_t)(k0 + sk + kk) * FD + sn];
                hi[kk] = (_Float16)B[(size_t)(k0 + sk + kk + 8) * FD + sn];
            }
            *(v8h*)&Bs[sn][sk]     = lo;   // 2x ds_store_b128 total
            *(v8h*)&Bs[sn][sk + 8] = hi;
        }
        __syncthreads();

        if (k0 + 32 < K) __builtin_prefetch(Arow + k0 + 32, 0, 0);

        // ---- A fragment (16x32 f16): lanes 0-15 hold K {0..7,16..23},
        //      lanes 16-31 hold K {8..15,24..31} of this step. ----
        const float* p0 = Arow + k0 + half * 8;        // 16-B aligned
        const float* p1 = Arow + k0 + 16 + half * 8;
        v4f g0 = rowok ? *(const v4f*)(p0)     : (v4f){};
        v4f g1 = rowok ? *(const v4f*)(p0 + 4) : (v4f){};
        v4f g2 = rowok ? *(const v4f*)(p1)     : (v4f){};
        v4f g3 = rowok ? *(const v4f*)(p1 + 4) : (v4f){};
        v16h a;
        #pragma unroll
        for (int e = 0; e < 4; ++e) {
            a[e]      = (_Float16)g0[e];
            a[e + 4]  = (_Float16)g1[e];
            a[e + 8]  = (_Float16)g2[e];
            a[e + 12] = (_Float16)g3[e];
        }

        // ---- load all 8 B fragments first (16 back-to-back ds_load_b128),
        //      then issue the 8 WMMAs so DS latency hides under matrix issue
        v16h bfrag[8];
        #pragma unroll
        for (int j = 0; j < 8; ++j) {
            const _Float16* bp = &Bs[j * 16 + lrow][half * 8];
            v8h blo = *(const v8h*)(bp);
            v8h bhi = *(const v8h*)(bp + 16);
            bfrag[j] = __builtin_shufflevector(blo, bhi,
                0, 1, 2, 3, 4, 5, 6, 7, 8, 9, 10, 11, 12, 13, 14, 15);
        }
        #pragma unroll
        for (int j = 0; j < 8; ++j) {
            acc[j] = __builtin_amdgcn_wmma_f32_16x16x32_f16(
                false, a, false, bfrag[j], (short)0, acc[j], false, false);
        }
        __syncthreads();
    }

    // C/D layout: vgpr r, lanes 0-15 -> M=r, lanes 16-31 -> M=r+8; N = lane%16
    #pragma unroll
    for (int j = 0; j < 8; ++j) {
        int col = j * 16 + lrow;
        float bv = bias ? bias[col] : 0.0f;
        #pragma unroll
        for (int r = 0; r < 8; ++r) {
            int row = rowbase + r + half * 8;
            if (row < M) {
                float v = acc[j][r] + bv;
                if (act == 1) v = v > 0.0f ? v : 0.01f * v;
                C[(size_t)row * ldc + coloff + col] = v;
            }
        }
    }
}

// ---------------------------------------------------------------------------
// value projection: vt[n, 0:128] = value[n,:] (N x 12) @ fc1_w (12 x 128) + b
// tiny K -> plain VALU kernel, one block per node, thread = output feature
// ---------------------------------------------------------------------------
__global__ __launch_bounds__(128)
void shgn_value_fc(const float* __restrict__ value, const float* __restrict__ w,
                   const float* __restrict__ b, float* __restrict__ vt)
{
    int n = blockIdx.x;
    int f = threadIdx.x;
    const float* row = value + (size_t)n * VN;
    float acc = b[f];
    #pragma unroll
    for (int k = 0; k < VN; ++k)
        acc += row[k] * w[k * FD + f];
    vt[(size_t)n * 256 + f] = acc;
}

// ---------------------------------------------------------------------------
// per-node attention scores: s_i[n] = Wx[n,:]·a_i ; s_j[n] = Wx[n,:]·a_j
// ---------------------------------------------------------------------------
__global__ __launch_bounds__(256)
void shgn_node_scores(const float* __restrict__ Wx, const float* __restrict__ avec,
                      int M, float* __restrict__ s_i, float* __restrict__ s_j)
{
    int wave = threadIdx.x >> 5, lane = threadIdx.x & 31;
    int n = blockIdx.x * 8 + wave;
    if (n >= M) return;
    const float* row = Wx + (size_t)n * FD;
    float si = 0.0f, sj = 0.0f;
    #pragma unroll
    for (int f = lane; f < FD; f += 32) {
        float x = row[f];
        si += x * avec[f];
        sj += x * avec[FD + f];
    }
    #pragma unroll
    for (int off = 16; off > 0; off >>= 1) {
        si += __shfl_xor(si, off);
        sj += __shfl_xor(sj, off);
    }
    if (lane == 0) { s_i[n] = si; s_j[n] = sj; }
}

// s_r = rel_emb(3x200) @ Wr(200x128) @ a_r(128)
__global__ __launch_bounds__(256)
void shgn_rel_scores(const float* __restrict__ rel, const float* __restrict__ Wr,
                     const float* __restrict__ a_r, float* __restrict__ s_r)
{
    __shared__ float w[REL];
    for (int r = threadIdx.x; r < REL; r += 256) {
        float s = 0.0f;
        for (int f = 0; f < FD; ++f) s += Wr[(size_t)r * FD + f] * a_r[f];
        w[r] = s;
    }
    __syncthreads();
    if (threadIdx.x < NET) {
        float s = 0.0f;
        for (int r = 0; r < REL; ++r) s += rel[threadIdx.x * REL + r] * w[r];
        s_r[threadIdx.x] = s;
    }
}

// e = leaky_relu(s_i[dst]+s_j[src]+s_r[type], 0.2); segment max via atomicMax
__global__ void shgn_edge_logits(const int* __restrict__ ei, const int* __restrict__ et,
                                 const float* __restrict__ s_i, const float* __restrict__ s_j,
                                 const float* __restrict__ s_r,
                                 float* __restrict__ ebuf, unsigned* __restrict__ mkey, int E)
{
    int e = blockIdx.x * blockDim.x + threadIdx.x;
    if (e >= E) return;
    int src = ei[e], dst = ei[E + e];
    float v = s_i[dst] + s_j[src] + s_r[et[e]];
    v = v > 0.0f ? v : 0.2f * v;
    ebuf[e] = v;
    atomicMax(&mkey[dst], f2key(v));
}

// ex = exp(e - m[dst]); denom = segment_sum(ex)
__global__ void shgn_edge_expsum(const int* __restrict__ ei, const float* __restrict__ ein,
                                 const unsigned* __restrict__ mkey,
                                 float* __restrict__ exbuf, float* __restrict__ denom, int E)
{
    int e = blockIdx.x * blockDim.x + threadIdx.x;
    if (e >= E) return;
    int dst = ei[E + e];
    float ex = expf(ein[e] - key2f(mkey[dst]));
    exbuf[e] = ex;
    atomicAdd(&denom[dst], ex);
}

// agg[dst,:] += Wx[src,:] * (ex/denom[dst]) — 128 threads per edge
__global__ __launch_bounds__(256)
void shgn_edge_scatter(const int* __restrict__ ei, const float* __restrict__ exbuf,
                       const float* __restrict__ denom, const float* __restrict__ Wx,
                       float* __restrict__ agg, int E)
{
    int eo = threadIdx.x >> 7;
    int f  = threadIdx.x & 127;
    int e = blockIdx.x * 2 + eo;
    if (e >= E) return;
    int src = ei[e], dst = ei[E + e];
    float alpha = exbuf[e] / denom[dst];
    atomicAdd(&agg[(size_t)dst * FD + f], Wx[(size_t)src * FD + f] * alpha);
}

// out = elu(agg + xres), in-place on agg
__global__ void shgn_elu_combine(float* __restrict__ agg, const float* __restrict__ xres,
                                 size_t n)
{
    size_t i = blockIdx.x * (size_t)blockDim.x + threadIdx.x;
    if (i >= n) return;
    float v = agg[i] + xres[i];
    agg[i] = v > 0.0f ? v : (expf(v) - 1.0f);
}

// out[i,:] = feat[idx[i],:] @ fc3_w(128x2) + fc3_b
__global__ void shgn_final_gather(const float* __restrict__ feat, const int* __restrict__ idx,
                                  const float* __restrict__ w, const float* __restrict__ b,
                                  float* __restrict__ out, int n)
{
    int i = blockIdx.x * blockDim.x + threadIdx.x;
    if (i >= n) return;
    const float* row = feat + (size_t)idx[i] * FD;
    float a0 = b[0], a1 = b[1];
    for (int f = 0; f < FD; ++f) {
        float x = row[f];
        a0 += x * w[f * 2 + 0];
        a1 += x * w[f * 2 + 1];
    }
    out[i * 2 + 0] = a0;
    out[i * 2 + 1] = a1;
}

// ---------------------------------------------------------------------------
extern "C" void kernel_launch(void* const* d_in, const int* in_sizes, int n_in,
                              void* d_out, int out_size, void* d_ws, size_t ws_size,
                              hipStream_t stream)
{
    (void)in_sizes; (void)n_in; (void)out_size; (void)ws_size;

    const float* value  = (const float*)d_in[0];
    const float* text   = (const float*)d_in[1];
    const float* fc1_w  = (const float*)d_in[2];
    const float* fc1_b  = (const float*)d_in[3];
    const float* fc2_w  = (const float*)d_in[4];
    const float* fc2_b  = (const float*)d_in[5];
    const float* relu_w = (const float*)d_in[6];
    const float* relu_b = (const float*)d_in[7];
    const float* l1_W   = (const float*)d_in[8];
    const float* l1_Wr  = (const float*)d_in[9];
    const float* l1_a   = (const float*)d_in[10];
    const float* l1_Wres= (const float*)d_in[11];
    const float* l1_rel = (const float*)d_in[12];
    const float* l2_W   = (const float*)d_in[13];
    const float* l2_Wr  = (const float*)d_in[14];
    const float* l2_a   = (const float*)d_in[15];
    const float* l2_Wres= (const float*)d_in[16];
    const float* l2_rel = (const float*)d_in[17];
    const float* fc3_w  = (const float*)d_in[18];
    const float* fc3_b  = (const float*)d_in[19];
    const int*   ei     = (const int*)d_in[20];   // [2, E]: src row 0, dst row 1
    const int*   et     = (const int*)d_in[21];
    const int*   idx    = (const int*)d_in[22];
    float* out = (float*)d_out;

    // workspace carve-out (~105 MB)
    char* base = (char*)d_ws;
    size_t off = 0;
    auto take = [&](size_t bytes) -> char* {
        char* p = base + off;
        off = (off + bytes + 255) & ~(size_t)255;
        return p;
    };
    float*    vt    = (float*)take((size_t)N_NODES * 256 * 4);   // concat buf / Wx+Xres
    float*    featA = (float*)take((size_t)N_NODES * FD * 4);
    float*    featB = (float*)take((size_t)N_NODES * FD * 4);
    float*    s_i   = (float*)take((size_t)N_NODES * 4);
    float*    s_j   = (float*)take((size_t)N_NODES * 4);
    unsigned* mkey  = (unsigned*)take((size_t)N_NODES * 4);
    float*    denom = (float*)take((size_t)N_NODES * 4);
    float*    exbuf = (float*)take((size_t)E_EDGES * 4);
    float*    s_r   = (float*)take(64);
    float* Wx   = vt;                       // reuse concat space after encoder
    float* Xres = vt + (size_t)N_NODES * FD;

    const int gemmGrid = (N_NODES + 127) / 128;

    // --- encoder: feat = leaky_relu([v | t] @ relu_w + b, 0.01) ---
    shgn_value_fc<<<N_NODES, 128, 0, stream>>>(value, fc1_w, fc1_b, vt);
    shgn_gemm_x128<<<gemmGrid, 256, 0, stream>>>(text, TN, N_NODES, TN,
                                                 fc2_w, fc2_b, vt, 256, FD, 0);
    shgn_gemm_x128<<<gemmGrid, 256, 0, stream>>>(vt, 256, N_NODES, 256,
                                                 relu_w, relu_b, featA, FD, 0, 1);

    auto layer = [&](const float* x, float* o, const float* W, const float* Wr,
                     const float* a, const float* Wres, const float* rel) {
        shgn_gemm_x128<<<gemmGrid, 256, 0, stream>>>(x, FD, N_NODES, FD,
                                                     W, nullptr, Wx, FD, 0, 0);
        shgn_gemm_x128<<<gemmGrid, 256, 0, stream>>>(x, FD, N_NODES, FD,
                                                     Wres, nullptr, Xres, FD, 0, 0);
        shgn_node_scores<<<(N_NODES + 7) / 8, 256, 0, stream>>>(Wx, a, N_NODES, s_i, s_j);
        shgn_rel_scores<<<1, 256, 0, stream>>>(rel, Wr, a + 2 * FD, s_r);
        hipMemsetAsync(mkey, 0, (size_t)N_NODES * 4, stream);
        hipMemsetAsync(denom, 0, (size_t)N_NODES * 4, stream);
        shgn_edge_logits<<<(E_EDGES + 255) / 256, 256, 0, stream>>>(
            ei, et, s_i, s_j, s_r, exbuf, mkey, E_EDGES);
        shgn_edge_expsum<<<(E_EDGES + 255) / 256, 256, 0, stream>>>(
            ei, exbuf, mkey, exbuf, denom, E_EDGES);
        hipMemsetAsync(o, 0, (size_t)N_NODES * FD * 4, stream);
        shgn_edge_scatter<<<(E_EDGES + 1) / 2, 256, 0, stream>>>(
            ei, exbuf, denom, Wx, o, E_EDGES);
        shgn_elu_combine<<<(int)(((size_t)N_NODES * FD + 255) / 256), 256, 0, stream>>>(
            o, Xres, (size_t)N_NODES * FD);
    };

    layer(featA, featB, l1_W, l1_Wr, l1_a, l1_Wres, l1_rel);
    layer(featB, featA, l2_W, l2_Wr, l2_a, l2_Wres, l2_rel);

    shgn_final_gather<<<(NIDX + 255) / 256, 256, 0, stream>>>(
        featA, idx, fc3_w, fc3_b, out, NIDX);
}
